// InternalNode_66743791780081
// MI455X (gfx1250) — matrix operand order repaired and checked by reading, not compiled, not run on hardware
//
#include <hip/hip_runtime.h>

// ---------------------------------------------------------------------------
// Soft decision tree inference, reformulated as:
//   W[b,l]  = exp( sum of edge log-probs on path root->leaf l )   (fp16)
//   ST[c,l] = softmax(leaf_logits[l])[c]                          (fp16, transposed, N padded)
//   out[b,c]= log( sum_l W[b,l] * ST[c,l] )  -> WMMA GEMM, fp32 accumulate
// ---------------------------------------------------------------------------

#define DEPTH       8
#define BATCH       2048
#define LEAVES      256        // 2^DEPTH
#define NODES       255        // 2^DEPTH - 1
#define NCLASS      200
#define NPAD        208        // 13 tiles of 16
#define KDIM        256        // = LEAVES
#define EPSV        1e-7f
#define LN2         0.6931471805599453f

typedef __attribute__((ext_vector_type(16))) _Float16 v16h;
typedef __attribute__((ext_vector_type(8)))  _Float16 v8h;
typedef __attribute__((ext_vector_type(8)))  float    v8f;

// ---------------------------------------------------------------------------
// Kernel 1: per-leaf softmax, written transposed into ST[NPAD][KDIM] (fp16).
// Columns 200..207 are zero padding (contribute nothing to the GEMM).
// One 256-thread block per leaf.
// ---------------------------------------------------------------------------
__global__ __launch_bounds__(256) void leaf_softmax_kernel(
    const float* __restrict__ leaf_logits,   // [LEAVES][NCLASS]
    _Float16* __restrict__ ST)               // [NPAD][KDIM]
{
  const int l = blockIdx.x;       // leaf 0..255
  const int c = threadIdx.x;      // class slot 0..255
  __shared__ float red[256];

  float x = (c < NCLASS) ? leaf_logits[l * NCLASS + c] : -INFINITY;

  // max-reduce
  red[c] = x;
  __syncthreads();
  for (int s = 128; s > 0; s >>= 1) {
    if (c < s) red[c] = fmaxf(red[c], red[c + s]);
    __syncthreads();
  }
  const float m = red[0];
  __syncthreads();

  // sum-reduce of exp(x - m)
  const float e = (c < NCLASS) ? __expf(x - m) : 0.0f;
  red[c] = e;
  __syncthreads();
  for (int s = 128; s > 0; s >>= 1) {
    if (c < s) red[c] += red[c + s];
    __syncthreads();
  }
  const float inv = 1.0f / red[0];

  if (c < NPAD) ST[c * KDIM + l] = (_Float16)(e * inv);   // 0 for pad slots
}

// ---------------------------------------------------------------------------
// Kernel 2: path probabilities.  One thread per (batch, leaf): walk the heap
// from the root, bit d of the leaf index (MSB first) selects right/left.
// p_right = sims[node][b] (a log-prob <= 0); p_left via the numerically
// stable log(1 - exp(-a)) split at ln2, exactly as the reference.
// Threads: leaf index fastest -> coalesced W writes, broadcast sims reads.
// ---------------------------------------------------------------------------
__global__ __launch_bounds__(256) void path_weights_kernel(
    const float* __restrict__ sims,   // [NODES][BATCH]
    _Float16* __restrict__ W)         // [BATCH][KDIM]
{
  const int idx = blockIdx.x * blockDim.x + threadIdx.x;  // BATCH*LEAVES threads
  const int l = idx & (LEAVES - 1);
  const int b = idx >> DEPTH;

  float logw = 0.0f;
  int node = 0;
  #pragma unroll
  for (int d = DEPTH - 1; d >= 0; --d) {
    const float s = sims[node * BATCH + b];      // log p_right, <= 0
    const int right = (l >> d) & 1;
    if (right) {
      logw += s;
    } else {
      const float a = fabsf(s) + EPSV;
      const float pl = (a < LN2) ? logf(-expm1f(-a)) : log1pf(-expf(-a));
      logw += pl;
    }
    node = 2 * node + 1 + right;
  }
  W[b * KDIM + l] = (_Float16)expf(logw);
}

// ---------------------------------------------------------------------------
// Kernel 3: tiled GEMM via v_wmma_f32_16x16x32_f16.
// One wave (32 threads) per 16x16 output tile; K = 256 in 8 steps of 32.
//   A fragment (16x32): lane<16 -> M=lane, K in {k0..k0+15};
//                       lane>=16 -> K in {k0+16..k0+31}.
//   B fragment (32x16): lane holds col=lane&15, 16 contiguous K values,
//                       upper half-wave offset by +16 in K (ST is transposed
//                       so these are contiguous b128 loads).
//   C/D: VGPR r -> row = r + 8*(lane>=16), col = lane&15.
// ---------------------------------------------------------------------------
__global__ __launch_bounds__(32) void tree_gemm_kernel(
    const _Float16* __restrict__ W,    // [BATCH][KDIM]
    const _Float16* __restrict__ ST,   // [NPAD][KDIM]
    float* __restrict__ out)           // [BATCH][NCLASS]
{
  const int lane  = threadIdx.x;       // 0..31
  const int tileM = blockIdx.x;        // 0..127
  const int tileN = blockIdx.y;        // 0..12
  const int half  = lane >> 4;         // 0 | 1
  const int lid   = lane & 15;

  const _Float16* pA = W  + (tileM * 16 + lid) * KDIM + half * 16;
  const _Float16* pB = ST + (tileN * 16 + lid) * KDIM + half * 16;

  v8f acc = {};
  #pragma unroll
  for (int kt = 0; kt < KDIM / 32; ++kt) {
    __builtin_prefetch(pA + 32, 0, 0);             // global_prefetch_b8
    const v8h a_lo = *(const v8h*)(pA);            // K k0   .. k0+7
    const v8h a_hi = *(const v8h*)(pA + 8);        // K k0+8 .. k0+15
    const v8h b_lo = *(const v8h*)(pB);            // K k0   .. k0+7
    const v8h b_hi = *(const v8h*)(pB + 8);        // K k0+8 .. k0+15
    const v16h a = __builtin_shufflevector(a_lo, a_hi,
        0,1,2,3,4,5,6,7,8,9,10,11,12,13,14,15);
    const v16h b = __builtin_shufflevector(b_lo, b_hi,
        0,1,2,3,4,5,6,7,8,9,10,11,12,13,14,15);
    acc = __builtin_amdgcn_wmma_f32_16x16x32_f16(
        /*neg_a=*/false, a, /*neg_b=*/false, b,
        /*c_mod=*/(short)0, acc, /*reuse_a=*/false, /*reuse_b=*/false);
    pA += 32;
    pB += 32;
  }

  const int col = tileN * 16 + lid;
  if (col < NCLASS) {
    #pragma unroll
    for (int r = 0; r < 8; ++r) {
      const int row = tileM * 16 + r + half * 8;
      out[row * NCLASS + col] = logf(acc[r]);
    }
  }
}

// ---------------------------------------------------------------------------
// Host launcher.  d_in[0] = similarities [255,2048] f32,
//                 d_in[1] = leaf_logits  [256,200]  f32,
//                 d_out   = pred_logits  [2048,200] f32.
// Workspace: W (fp16, 1 MiB) then ST (fp16, 104 KiB).
// ---------------------------------------------------------------------------
extern "C" void kernel_launch(void* const* d_in, const int* in_sizes, int n_in,
                              void* d_out, int out_size, void* d_ws, size_t ws_size,
                              hipStream_t stream) {
  (void)in_sizes; (void)n_in; (void)out_size; (void)ws_size;

  const float* sims        = (const float*)d_in[0];
  const float* leaf_logits = (const float*)d_in[1];
  float*       out         = (float*)d_out;

  _Float16* W  = (_Float16*)d_ws;                                        // [2048][256]
  _Float16* ST = (_Float16*)((char*)d_ws + (size_t)BATCH * KDIM * 2);    // [208][256]

  leaf_softmax_kernel<<<LEAVES, 256, 0, stream>>>(leaf_logits, ST);
  path_weights_kernel<<<(BATCH * LEAVES) / 256, 256, 0, stream>>>(sims, W);
  tree_gemm_kernel<<<dim3(BATCH / 16, NPAD / 16), 32, 0, stream>>>(W, ST, out);
}
